// CausalSelfAttention_51084341018984
// MI455X (gfx1250) — matrix operand order repaired and checked by reading, not compiled, not run on hardware
//
#include <hip/hip_runtime.h>

#define B_   2
#define T_   2048
#define C_   1024
#define H_   16
#define HD_  64
#define C3_  3072

typedef __attribute__((ext_vector_type(16))) __bf16 v16bf;
typedef __attribute__((ext_vector_type(8)))  float  v8f;

union BF16x16 { v16bf v; unsigned short s[16]; };

__device__ __forceinline__ unsigned short f2bf(float f) {
    unsigned int u = __float_as_uint(f);
    unsigned int r = u + 0x7FFFu + ((u >> 16) & 1u);   // round-to-nearest-even
    return (unsigned short)(r >> 16);
}

// ---- packed fragment layout ------------------------------------------------
// One fragment block = 512 bf16 (1 KB): lane L owns halves [L*16, L*16+16).
__device__ __forceinline__ int packedA_index(int r, int kk) {
    int lane = (((kk >> 3) & 1) << 4) | r;
    int slot = (kk & 7) | ((kk >> 4) << 3);
    return lane * 16 + slot;
}
__device__ __forceinline__ int packedB_index(int r, int c) {
    int lane = (((r >> 3) & 1) << 4) | c;
    int slot = (r & 7) | ((r >> 4) << 3);
    return lane * 16 + slot;
}

// Load one packed fragment: 32 contiguous bytes per lane (2 x b128).
__device__ __forceinline__ v16bf load_packed(const unsigned short* blk) {
    const unsigned short* p = blk + (threadIdx.x & 31) * 16;
    BF16x16 u;
    __builtin_memcpy(&u.s[0], p, 16);
    __builtin_memcpy(&u.s[8], p + 8, 16);
    return u.v;
}

// A fragment from LDS, row-major 16x32 tile, stride 40 halves (16B-aligned chunks)
#define PSTRIDE 40
__device__ __forceinline__ v16bf load_a_lds(const unsigned short* base) {
    int lane = threadIdx.x & 31;
    const unsigned short* p = base + (lane & 15) * PSTRIDE + ((lane >> 4) << 3);
    BF16x16 u;
    __builtin_memcpy(&u.s[0], p, 16);
    __builtin_memcpy(&u.s[8], p + 16, 16);
    return u.v;
}

__device__ __forceinline__ v8f wmma_bf16(v16bf a, v16bf b, v8f c) {
    return __builtin_amdgcn_wmma_f32_16x16x32_bf16(false, a, false, b, (short)0, c, false, false);
}

// ---------------------------------------------------------------- pack kernels
__global__ void k_pack_a(const float* __restrict__ in, unsigned short* __restrict__ out,
                         int M, int Kd) {
    int e = blockIdx.x * blockDim.x + threadIdx.x;
    if (e >= M * Kd) return;
    int m = e / Kd, k = e % Kd;
    int blk = (m >> 4) * (Kd >> 5) + (k >> 5);
    out[blk * 512 + packedA_index(m & 15, k & 31)] = f2bf(in[e]);
}

__global__ void k_pack_b(const float* __restrict__ in, unsigned short* __restrict__ out,
                         int Kd, int N) {
    int e = blockIdx.x * blockDim.x + threadIdx.x;
    if (e >= Kd * N) return;
    int k = e / N, n = e % N;
    int blk = (k >> 5) * (N >> 4) + (n >> 4);     // K-outer, N-inner block order
    out[blk * 512 + packedB_index(k & 31, n & 15)] = f2bf(in[e]);
}

// ---------------------------------------------------------------- QKV GEMM
// qkv = X[4096x1024] @ W[1024x3072] + b_attn, wave computes 32(M)x64(N):
// 2 A-frags + 4 B-frags -> 8 WMMAs per K-step. launch_bounds(256,1) so the
// 64 accumulator VGPRs + pipelined fragments fit without scratch spills.
__global__ void __launch_bounds__(256, 1)
k_qkv(const unsigned short* __restrict__ Xp,
      const unsigned short* __restrict__ Wp,
      const float* __restrict__ bias,
      unsigned short* __restrict__ Qp,
      unsigned short* __restrict__ Ktp,
      unsigned short* __restrict__ Vp) {
    int wave = blockIdx.x * (blockDim.x >> 5) + (threadIdx.x >> 5);
    const int NT64 = C3_ / 64;                     // 48 col-64 tiles
    int tm = wave / NT64, tn = wave % NT64;        // tm in [0,128): 32-row stripe
    int mt0 = tm * 2, mt1 = tm * 2 + 1;            // 16-row tile indices

    const unsigned short* ap0 = Xp + (size_t)mt0 * (C_ / 32) * 512;
    const unsigned short* ap1 = Xp + (size_t)mt1 * (C_ / 32) * 512;

    v8f c00 = {}, c01 = {}, c02 = {}, c03 = {};
    v8f c10 = {}, c11 = {}, c12 = {}, c13 = {};
    for (int kt = 0; kt < C_ / 32; ++kt) {
        const unsigned short* wb = Wp + (size_t)(kt * (C3_ / 16) + tn * 4) * 512;
        if (kt + 1 < C_ / 32)
            __builtin_prefetch(wb + (size_t)(C3_ / 16) * 512, 0, 1);
        v16bf a0 = load_packed(ap0 + (size_t)kt * 512);
        v16bf a1 = load_packed(ap1 + (size_t)kt * 512);
        v16bf b;
        b = load_packed(wb + 0 * 512); c00 = wmma_bf16(a0, b, c00); c10 = wmma_bf16(a1, b, c10);
        b = load_packed(wb + 1 * 512); c01 = wmma_bf16(a0, b, c01); c11 = wmma_bf16(a1, b, c11);
        b = load_packed(wb + 2 * 512); c02 = wmma_bf16(a0, b, c02); c12 = wmma_bf16(a1, b, c12);
        b = load_packed(wb + 3 * 512); c03 = wmma_bf16(a0, b, c03); c13 = wmma_bf16(a1, b, c13);
    }
    int lane = threadIdx.x & 31;
    int col  = lane & 15, half = lane >> 4;

    auto scatter = [&](const v8f& acc, int j, int mi) {
        int n = tn * 64 + j * 16 + col;
        float bv = bias[n];
        int which = n >> 10;                       // 0=q 1=k 2=v
        int cc = n & (C_ - 1);
        int h = cc >> 6, d = cc & 63;
#pragma unroll
        for (int i = 0; i < 8; ++i) {
            int m = (mt0 + mi) * 16 + i + 8 * half;   // global token row
            int bb = m >> 11, t = m & (T_ - 1);
            int bh = bb * H_ + h;
            unsigned short val = f2bf(acc[i] + bv);
            if (which == 0) {
                size_t blk = (size_t)(bh * 128 + (t >> 4)) * 2 + (d >> 5);
                Qp[blk * 512 + packedA_index(t & 15, d & 31)] = val;
            } else if (which == 1) {
                size_t blk = (size_t)(bh * 2 + (d >> 5)) * 128 + (t >> 4);
                Ktp[blk * 512 + packedB_index(d & 31, t & 15)] = val;
            } else {
                size_t blk = (size_t)(bh * 64 + (t >> 5)) * 4 + (d >> 4);
                Vp[blk * 512 + packedB_index(t & 31, d & 15)] = val;
            }
        }
    };
    scatter(c00, 0, 0); scatter(c01, 1, 0); scatter(c02, 2, 0); scatter(c03, 3, 0);
    scatter(c10, 0, 1); scatter(c11, 1, 1); scatter(c12, 2, 1); scatter(c13, 3, 1);
}

// ---------------------------------------------------------------- flash attention
__global__ void __launch_bounds__(256, 1)
k_attn(const unsigned short* __restrict__ Qp,
       const unsigned short* __restrict__ Ktp,
       const unsigned short* __restrict__ Vp,
       unsigned short* __restrict__ Ybp) {
    __shared__ __align__(16) unsigned short ldsP[8][16 * PSTRIDE];
    int wslot = threadIdx.x >> 5;
    int wave  = blockIdx.x * (blockDim.x >> 5) + wslot;
    int qb = wave & 127;
    int h  = (wave >> 7) & 15;
    int b  = wave >> 11;
    int bh = b * H_ + h;
    int lane = threadIdx.x & 31;
    int col  = lane & 15, half = lane >> 4;

    const unsigned short* Qbase  = Qp  + (size_t)(bh * 128 + qb) * 2 * 512;
    const unsigned short* Ktbase = Ktp + (size_t)(bh * 2) * 128 * 512;
    const unsigned short* Vbase  = Vp  + (size_t)(bh * 64) * 4 * 512;

    v16bf aq0 = load_packed(Qbase);                // Q rows, K = d 0..31
    v16bf aq1 = load_packed(Qbase + 512);          // Q rows, K = d 32..63

    float mrow[8], lrow[8];
    v8f o0 = {}, o1 = {}, o2 = {}, o3 = {};
#pragma unroll
    for (int i = 0; i < 8; ++i) { mrow[i] = -__builtin_inff(); lrow[i] = 0.f; }

    int nkeys = qb * 16 + 16;
    for (int k0 = 0; k0 < nkeys; k0 += 32) {
        int nt0 = k0 >> 4;                         // key 16-block index
        v8f s0 = {}, s1 = {};
        s0 = wmma_bf16(aq0, load_packed(Ktbase + (size_t)(0 * 128 + nt0) * 512), s0);
        s0 = wmma_bf16(aq1, load_packed(Ktbase + (size_t)(1 * 128 + nt0) * 512), s0);
        s1 = wmma_bf16(aq0, load_packed(Ktbase + (size_t)(0 * 128 + nt0 + 1) * 512), s1);
        s1 = wmma_bf16(aq1, load_packed(Ktbase + (size_t)(1 * 128 + nt0 + 1) * 512), s1);

        const float scale = 0.125f;                // 1/sqrt(64)
#pragma unroll
        for (int i = 0; i < 8; ++i) {
            int q = qb * 16 + i + 8 * half;
            float v0 = s0[i] * scale;
            float v1 = s1[i] * scale;
            if (k0 + col > q)      v0 = -__builtin_inff();
            if (k0 + 16 + col > q) v1 = -__builtin_inff();
            s0[i] = v0; s1[i] = v1;
        }
#pragma unroll
        for (int i = 0; i < 8; ++i) {
            float mb = fmaxf(s0[i], s1[i]);
#pragma unroll
            for (int off = 8; off > 0; off >>= 1) mb = fmaxf(mb, __shfl_xor(mb, off));
            float mnew  = fmaxf(mrow[i], mb);
            float alpha = __expf(mrow[i] - mnew);
            float p0 = __expf(s0[i] - mnew);
            float p1 = __expf(s1[i] - mnew);
            float rs = p0 + p1;
#pragma unroll
            for (int off = 8; off > 0; off >>= 1) rs += __shfl_xor(rs, off);
            lrow[i] = lrow[i] * alpha + rs;
            mrow[i] = mnew;
            o0[i] *= alpha; o1[i] *= alpha; o2[i] *= alpha; o3[i] *= alpha;
            int r = i + 8 * half;                  // D-layout -> row-major P in LDS
            ldsP[wslot][r * PSTRIDE + col]      = f2bf(p0);
            ldsP[wslot][r * PSTRIDE + 16 + col] = f2bf(p1);
        }
        asm volatile("s_wait_dscnt 0" ::: "memory");   // same-wave LDS RAW
        v16bf pa = load_a_lds(&ldsP[wslot][0]);
        const unsigned short* vb = Vbase + (size_t)(k0 >> 5) * 4 * 512;
        o0 = wmma_bf16(pa, load_packed(vb + 0 * 512), o0);
        o1 = wmma_bf16(pa, load_packed(vb + 1 * 512), o1);
        o2 = wmma_bf16(pa, load_packed(vb + 2 * 512), o2);
        o3 = wmma_bf16(pa, load_packed(vb + 3 * 512), o3);
    }
    v8f oo[4] = { o0, o1, o2, o3 };
#pragma unroll
    for (int j = 0; j < 4; ++j) {
        int cgl = h * HD_ + j * 16 + col;
        int kt = cgl >> 5, kk = cgl & 31;
#pragma unroll
        for (int i = 0; i < 8; ++i) {
            int r = i + 8 * half;
            int mt = b * 128 + qb;                 // 16-row tile index
            float inv = 1.f / lrow[i];
            Ybp[((size_t)(mt * (C_ / 32) + kt)) * 512 + packedA_index(r, kk)] =
                f2bf(oo[j][i] * inv);
        }
    }
}

// ---------------------------------------------------------------- output projection
__global__ void __launch_bounds__(256, 1)
k_proj(const unsigned short* __restrict__ Ybp,
       const unsigned short* __restrict__ Wp,
       const float* __restrict__ bias,
       float* __restrict__ out) {
    int wave = blockIdx.x * (blockDim.x >> 5) + (threadIdx.x >> 5);
    const int NT64 = C_ / 64;                      // 16
    int tm = wave / NT64, tn = wave % NT64;
    int mt0 = tm * 2, mt1 = tm * 2 + 1;

    const unsigned short* ap0 = Ybp + (size_t)mt0 * (C_ / 32) * 512;
    const unsigned short* ap1 = Ybp + (size_t)mt1 * (C_ / 32) * 512;

    v8f c00 = {}, c01 = {}, c02 = {}, c03 = {};
    v8f c10 = {}, c11 = {}, c12 = {}, c13 = {};
    for (int kt = 0; kt < C_ / 32; ++kt) {
        const unsigned short* wb = Wp + (size_t)(kt * (C_ / 16) + tn * 4) * 512;
        if (kt + 1 < C_ / 32)
            __builtin_prefetch(wb + (size_t)(C_ / 16) * 512, 0, 1);
        v16bf a0 = load_packed(ap0 + (size_t)kt * 512);
        v16bf a1 = load_packed(ap1 + (size_t)kt * 512);
        v16bf b;
        b = load_packed(wb + 0 * 512); c00 = wmma_bf16(a0, b, c00); c10 = wmma_bf16(a1, b, c10);
        b = load_packed(wb + 1 * 512); c01 = wmma_bf16(a0, b, c01); c11 = wmma_bf16(a1, b, c11);
        b = load_packed(wb + 2 * 512); c02 = wmma_bf16(a0, b, c02); c12 = wmma_bf16(a1, b, c12);
        b = load_packed(wb + 3 * 512); c03 = wmma_bf16(a0, b, c03); c13 = wmma_bf16(a1, b, c13);
    }
    int lane = threadIdx.x & 31;
    int col  = lane & 15, half = lane >> 4;

    auto emit = [&](const v8f& acc, int j, int mi) {
        int n = tn * 64 + j * 16 + col;
        float bv = bias[n];
#pragma unroll
        for (int i = 0; i < 8; ++i) {
            int m = (mt0 + mi) * 16 + i + 8 * half;
            out[(size_t)m * C_ + n] = acc[i] + bv;
        }
    };
    emit(c00, 0, 0); emit(c01, 1, 0); emit(c02, 2, 0); emit(c03, 3, 0);
    emit(c10, 0, 1); emit(c11, 1, 1); emit(c12, 2, 1); emit(c13, 3, 1);
}

extern "C" void kernel_launch(void* const* d_in, const int* in_sizes, int n_in,
                              void* d_out, int out_size, void* d_ws, size_t ws_size,
                              hipStream_t stream) {
    const float* x      = (const float*)d_in[0];
    const float* w_attn = (const float*)d_in[1];
    const float* b_attn = (const float*)d_in[2];
    const float* w_proj = (const float*)d_in[3];
    const float* b_proj = (const float*)d_in[4];
    float* out = (float*)d_out;

    char* ws = (char*)d_ws;
    unsigned short* Xp    = (unsigned short*)(ws + 0);          //  8 MB packed A
    unsigned short* Wattn = (unsigned short*)(ws + 8388608);    //  6 MB packed B
    unsigned short* Wproj = (unsigned short*)(ws + 14680064);   //  2 MB packed B
    unsigned short* Qp    = (unsigned short*)(ws + 16777216);   //  8 MB packed A tiles
    unsigned short* Ktp   = (unsigned short*)(ws + 25165824);   //  8 MB packed B tiles
    unsigned short* Vp    = (unsigned short*)(ws + 33554432);   //  8 MB packed B tiles
    unsigned short* Ybp   = (unsigned short*)(ws + 41943040);   //  8 MB packed A tiles

    int nx  = B_ * T_ * C_;        // 4M
    int nwa = C_ * C3_;            // 3M
    int nwp = C_ * C_;             // 1M
    k_pack_a<<<(nx  + 255) / 256, 256, 0, stream>>>(x,      Xp,    B_ * T_, C_);
    k_pack_b<<<(nwa + 255) / 256, 256, 0, stream>>>(w_attn, Wattn, C_, C3_);
    k_pack_b<<<(nwp + 255) / 256, 256, 0, stream>>>(w_proj, Wproj, C_, C_);

    // 128 row32-stripes x 48 col64-tiles = 6144 waves / 8 per block
    k_qkv <<<768, 256, 0, stream>>>(Xp, Wattn, b_attn, Qp, Ktp, Vp);
    // 2*16*128 = 4096 waves
    k_attn<<<512, 256, 0, stream>>>(Qp, Ktp, Vp, Ybp);
    // 128 x 16 = 2048 waves
    k_proj<<<256, 256, 0, stream>>>(Ybp, Wproj, b_proj, out);
}
